// MATCC_30099130810523
// MI455X (gfx1250) — matrix operand order repaired and checked by reading, not compiled, not run on hardware
//
#include <hip/hip_runtime.h>
#include <hip/hip_bf16.h>

typedef _Float16 h8   __attribute__((ext_vector_type(8)));
typedef _Float16 h16  __attribute__((ext_vector_type(16)));
typedef float    f8   __attribute__((ext_vector_type(8)));

#define BN    2048
#define TSEQ  8
#define CC    256
#define MM    (BN * TSEQ)        /* 16384 rows */
#define MC    ((size_t)MM * CC)  /* 4194304 elements */
#define DFEAT 158
#define DGATE 63
#define DH    128                /* S_NHEAD head dim */
#define NW    14                 /* 256x256 weight matrices */

// ---------------------------------------------------------------------------
// helpers
// ---------------------------------------------------------------------------
__device__ __forceinline__ h16 pack16(h8 lo, h8 hi) {
  h16 r;
#pragma unroll
  for (int i = 0; i < 8; ++i) { r[i] = lo[i]; r[i + 8] = hi[i]; }
  return r;
}

__device__ __forceinline__ float sigmoidf_(float x) {
  return 1.0f / (1.0f + __expf(-x));
}

// ---------------------------------------------------------------------------
// weight f32 -> f16 conversion (14 matrices of 256x256)
// ---------------------------------------------------------------------------
struct WPtrs { const float* p[NW]; };

__global__ __launch_bounds__(256) void wcvt_kernel(WPtrs w, _Float16* __restrict__ out) {
  const size_t idx = (size_t)blockIdx.x * 256 + threadIdx.x;
  const int mi = (int)(idx >> 16);
  const int e  = (int)(idx & 65535);
  out[idx] = (_Float16)w.p[mi][e];
}

// ---------------------------------------------------------------------------
// prologue: gate filter + conv + input linear + series decomposition -> h
// one block per batch element, 256 threads = output channels
// ---------------------------------------------------------------------------
__global__ __launch_bounds__(256)
void prologue_kernel(const float* __restrict__ x,
                     const float* __restrict__ ftw, const float* __restrict__ ftb,
                     const float* __restrict__ fcw, const float* __restrict__ fcb,
                     const float* __restrict__ fpw, const float* __restrict__ fpb,
                     const float* __restrict__ inw, const float* __restrict__ inb,
                     const float* __restrict__ dlsw, const float* __restrict__ dlsb,
                     const float* __restrict__ dltw, const float* __restrict__ dltb,
                     float* __restrict__ hout)
{
  __shared__ float sGate[5 * DGATE];
  __shared__ float sGout[DFEAT];
  __shared__ float sSrc[TSEQ][DFEAT];

  const int b = blockIdx.x;
  const int tid = threadIdx.x;
  const size_t xb = (size_t)b * TSEQ * 221;

  for (int i = tid; i < 5 * DGATE; i += 256) {
    const int k = i / DGATE, j = i % DGATE;
    sGate[i] = x[xb + (size_t)k * 221 + DFEAT + j];
  }
  __syncthreads();

  if (tid < DFEAT) {
    float acc = 0.0f;
    for (int k = 0; k < 5; ++k) {
      float g = ftb[tid];
      for (int j = 0; j < DGATE; ++j) g += sGate[k * DGATE + j] * ftw[tid * DGATE + j];
      acc += g * fcw[tid * 5 + k];
    }
    acc += fcb[tid];
    sGout[tid] = acc * fpw[0] + fpb[0];
  }
  __syncthreads();

  for (int i = tid; i < TSEQ * DFEAT; i += 256) {
    const int t = i / DFEAT, f = i % DFEAT;
    sSrc[t][f] = x[xb + (size_t)t * 221 + f] + sGout[f];
  }
  __syncthreads();

  const int c = tid;
  float hv[TSEQ];
#pragma unroll
  for (int t = 0; t < TSEQ; ++t) hv[t] = inb[c];
  for (int f = 0; f < DFEAT; ++f) {
    const float w = inw[c * DFEAT + f];
#pragma unroll
    for (int t = 0; t < TSEQ; ++t) hv[t] += sSrc[t][f] * w;
  }

  float tr[TSEQ], se[TSEQ];
#pragma unroll
  for (int t = 0; t < TSEQ; ++t) {
    const float a  = hv[t == 0 ? 0 : t - 1];
    const float m  = hv[t];
    const float cN = hv[t == TSEQ - 1 ? TSEQ - 1 : t + 1];
    tr[t] = (a + m + cN) * (1.0f / 3.0f);
    se[t] = hv[t] - tr[t];
  }
#pragma unroll
  for (int t = 0; t < TSEQ; ++t) {
    float acc = dlsb[t] + dltb[t];
    for (int u = 0; u < TSEQ; ++u) acc += se[u] * dlsw[t * TSEQ + u] + tr[u] * dltw[t * TSEQ + u];
    hout[((size_t)b * TSEQ + t) * CC + c] = acc;
  }
}

// ---------------------------------------------------------------------------
// layernorm over C=256 (optional 2nd input added first)
// ---------------------------------------------------------------------------
__global__ __launch_bounds__(256)
void layernorm_kernel(const float* __restrict__ in, const float* __restrict__ addp,
                      const float* __restrict__ g, const float* __restrict__ bta,
                      float* __restrict__ out)
{
  __shared__ float red[256];
  const int row = blockIdx.x;
  const int c = threadIdx.x;
  const size_t idx = (size_t)row * CC + c;
  float v = in[idx];
  if (addp) v += addp[idx];
  red[c] = v;
  __syncthreads();
  for (int s = 128; s > 0; s >>= 1) { if (c < s) red[c] += red[c + s]; __syncthreads(); }
  const float mean = red[0] * (1.0f / 256.0f);
  __syncthreads();
  const float d = v - mean;
  red[c] = d * d;
  __syncthreads();
  for (int s = 128; s > 0; s >>= 1) { if (c < s) red[c] += red[c + s]; __syncthreads(); }
  const float var = red[0] * (1.0f / 256.0f);
  out[idx] = d * rsqrtf(var + 1e-5f) * g[c] + bta[c];
}

// ---------------------------------------------------------------------------
// time-shift-half + convert to f16   (first 128 channels from t-1, zero at t=0)
// ---------------------------------------------------------------------------
__global__ __launch_bounds__(256)
void tshift_cvt_kernel(const float* __restrict__ in, _Float16* __restrict__ out)
{
  const size_t idx = (size_t)blockIdx.x * 256 + threadIdx.x;
  const int c = (int)(idx & 255);
  const int t = (int)((idx >> 8) & 7);
  float v;
  if (c < 128) v = (t == 0) ? 0.0f : in[idx - CC];
  else         v = in[idx];
  out[idx] = (_Float16)v;
}

__global__ __launch_bounds__(256)
void cvt_kernel(const float* __restrict__ in, _Float16* __restrict__ out)
{
  const size_t idx = (size_t)blockIdx.x * 256 + threadIdx.x;
  out[idx] = (_Float16)in[idx];
}

// ---------------------------------------------------------------------------
// WMMA GEMM: Y[M,256] = epi( X16[M,256] @ W16[256,256]^T + bias )
// one wave per 16x64 output strip; f16 WMMA 16x16x32, f32 accumulate.
// vstore==1: scatter f16 output in per-(t,h) transposed layout for attention V
// ---------------------------------------------------------------------------
template<int EPI>   // 0 none, 1 exp(min(.,30)), 2 relu
__global__ __launch_bounds__(32)
void gemm256_kernel(const _Float16* __restrict__ A,
                    const _Float16* __restrict__ W,
                    const float*    __restrict__ bias,
                    float*          __restrict__ outF,
                    _Float16*       __restrict__ outH,
                    int vstore)
{
  const int lane = threadIdx.x;
  const int lmod = lane & 15;
  const int lhi  = lane >> 4;           // 0 or 1
  const int m0 = blockIdx.x * 16;
  const int n0 = blockIdx.y * 64;

  const _Float16* arow = A + (size_t)(m0 + lmod) * CC;
  const _Float16* wrow[4];
#pragma unroll
  for (int j = 0; j < 4; ++j) wrow[j] = W + (size_t)(n0 + j * 16 + lmod) * CC;

  f8 acc[4] = {};
#pragma unroll
  for (int k0 = 0; k0 < CC; k0 += 32) {
    const h16 a = pack16(*(const h8*)(arow + k0 + lhi * 16),
                         *(const h8*)(arow + k0 + lhi * 16 + 8));
    h16 bf[4];
#pragma unroll
    for (int j = 0; j < 4; ++j)
      bf[j] = pack16(*(const h8*)(wrow[j] + k0 + lhi * 16),
                     *(const h8*)(wrow[j] + k0 + lhi * 16 + 8));
#pragma unroll
    for (int j = 0; j < 4; ++j)
      acc[j] = __builtin_amdgcn_wmma_f32_16x16x32_f16(false, a, false, bf[j],
                                                      (short)0, acc[j], false, false);
  }

#pragma unroll
  for (int j = 0; j < 4; ++j) {
    const int n = n0 + j * 16 + lmod;
    const float bv = bias ? bias[n] : 0.0f;
#pragma unroll
    for (int r = 0; r < 8; ++r) {
      const int m = m0 + r + lhi * 8;
      float v = acc[j][r] + bv;
      if (EPI == 1) v = __expf(fminf(v, 30.0f));
      if (EPI == 2) v = fmaxf(v, 0.0f);
      if (outF) outF[(size_t)m * CC + n] = v;
      if (outH) {
        if (vstore) {
          const int bb = m >> 3, tt = m & 7, hh = n >> 7, d = n & 127;
          outH[((size_t)(tt * 2 + hh) * DH + d) * BN + bb] = (_Float16)v;
        } else {
          outH[(size_t)m * CC + n] = (_Float16)v;
        }
      }
    }
  }
}

// ---------------------------------------------------------------------------
// timemix post: k=exp-space, v, r rows -> rwkv (f16) ; per (b, channel)
// ---------------------------------------------------------------------------
__global__ __launch_bounds__(256)
void timemix_post_kernel(const float* __restrict__ kk, const float* __restrict__ vv,
                         const float* __restrict__ rr,
                         const float* __restrict__ time_w, const float* __restrict__ time_a,
                         const float* __restrict__ time_b,
                         _Float16* __restrict__ outH)
{
  const int b = blockIdx.x;
  const int c = threadIdx.x;
  const int hh = c >> 6;                 // T_NHEAD=4, hs=64
  const size_t base = (size_t)b * TSEQ * CC + c;
  float kl[TSEQ], kv[TSEQ];
#pragma unroll
  for (int t = 0; t < TSEQ; ++t) {
    kl[t] = kk[base + (size_t)t * CC];
    kv[t] = kl[t] * vv[base + (size_t)t * CC];
  }
  float sumk = 0.0f;
#pragma unroll
  for (int t = 0; t < TSEQ; ++t) {
    sumk += kl[t];
    const float bt = time_b[hh * 300 + t];
    float acc = 0.0f;
    for (int u = 0; u <= t; ++u) {
      const float w = time_w[hh * 300 + 299 - t + u] * time_a[hh * 300 + u] * bt;
      acc += w * kv[u];
    }
    const float r = rr[base + (size_t)t * CC];
    outH[base + (size_t)t * CC] = (_Float16)(sigmoidf_(r) * acc / sumk);
  }
}

// ---------------------------------------------------------------------------
// elementwise kernels
// ---------------------------------------------------------------------------
__global__ __launch_bounds__(256)
void resid_gamma_kernel(float* __restrict__ h, const float* __restrict__ g,
                        const float* __restrict__ gamma)
{
  const size_t idx = (size_t)blockIdx.x * 256 + threadIdx.x;
  const int t = (int)((idx >> 8) & 7);
  h[idx] += g[idx] * gamma[t];
}

__global__ __launch_bounds__(256)
void mishmul_kernel(const float* __restrict__ k, const float* __restrict__ v,
                    _Float16* __restrict__ out)
{
  const size_t idx = (size_t)blockIdx.x * 256 + threadIdx.x;
  const float kx = k[idx];
  const float sp = (kx > 20.0f) ? kx : log1pf(__expf(kx));
  out[idx] = (_Float16)(kx * tanhf(sp) * v[idx]);
}

__global__ __launch_bounds__(256)
void chanmix_resid_kernel(float* __restrict__ h, const float* __restrict__ r,
                          const float* __restrict__ wkv)
{
  const size_t idx = (size_t)blockIdx.x * 256 + threadIdx.x;
  h[idx] += sigmoidf_(r[idx]) * wkv[idx];
}

__global__ __launch_bounds__(256)
void add_kernel(const float* __restrict__ a, const float* __restrict__ b,
                float* __restrict__ out)
{
  const size_t idx = (size_t)blockIdx.x * 256 + threadIdx.x;
  out[idx] = a[idx] + b[idx];
}

// ---------------------------------------------------------------------------
// flash attention over the batch axis: one wave per (q-tile of 16, t, h)
// Q,K: [b][t][256] f16 ;  VT: [(t*2+h)*128+d][b] f16 ;  att out f32 [b][t][256]
// All K fragments (2x4) and V fragments (8) for a 32-key tile are preloaded
// into distinct registers before the WMMA chains so loads overlap (no
// single-buffer s_wait_loadcnt serialization); V loads also overlap softmax.
// ---------------------------------------------------------------------------
__global__ __launch_bounds__(32)
void flash_attn_kernel(const _Float16* __restrict__ Q,
                       const _Float16* __restrict__ K,
                       const _Float16* __restrict__ VT,
                       float* __restrict__ att)
{
  __shared__ float    sS[16][32];
  __shared__ _Float16 sP[16][32];
  __shared__ float    sScale[16];
  __shared__ float    sSum[16];

  const int lane = threadIdx.x;
  const int lmod = lane & 15;
  const int lhi  = lane >> 4;
  const int m0 = blockIdx.x * 16;          // query tile (batch rows)
  const int th = blockIdx.y;               // t*2 + h
  const int t  = th >> 1;
  const int hh = th & 1;
  const float invT = 0.088388347648318447f;   // 1/sqrt(128)

  const _Float16* qrow = Q + ((size_t)(m0 + lmod) * TSEQ + t) * CC + hh * DH;
  h16 qf[4];
#pragma unroll
  for (int kk = 0; kk < 4; ++kk)
    qf[kk] = pack16(*(const h8*)(qrow + kk * 32 + lhi * 16),
                    *(const h8*)(qrow + kk * 32 + lhi * 16 + 8));

  f8 o[8] = {};
  float runmax = -1e30f, runsum = 0.0f;

  for (int j0 = 0; j0 < BN; j0 += 32) {
    // ---- issue all V-fragment loads for this key tile first (latency is
    //      covered by the S WMMAs + softmax below)
    h16 vf[8];
#pragma unroll
    for (int d = 0; d < 8; ++d) {
      const _Float16* vrow = VT + ((size_t)th * DH + d * 16 + lmod) * BN + j0 + lhi * 16;
      vf[d] = pack16(*(const h8*)(vrow), *(const h8*)(vrow + 8));
    }

    // ---- preload all 8 K fragments (2 j-tiles x 4 k-steps), then WMMA chain
    h16 kf[2][4];
#pragma unroll
    for (int jj = 0; jj < 2; ++jj) {
      const _Float16* krow = K + ((size_t)(j0 + jj * 16 + lmod) * TSEQ + t) * CC + hh * DH;
#pragma unroll
      for (int kk = 0; kk < 4; ++kk)
        kf[jj][kk] = pack16(*(const h8*)(krow + kk * 32 + lhi * 16),
                            *(const h8*)(krow + kk * 32 + lhi * 16 + 8));
    }

    f8 s[2] = {};
#pragma unroll
    for (int kk = 0; kk < 4; ++kk) {
#pragma unroll
      for (int jj = 0; jj < 2; ++jj)
        s[jj] = __builtin_amdgcn_wmma_f32_16x16x32_f16(false, qf[kk], false, kf[jj][kk],
                                                       (short)0, s[jj], false, false);
    }
#pragma unroll
    for (int jj = 0; jj < 2; ++jj)
#pragma unroll
      for (int r = 0; r < 8; ++r)
        sS[r + lhi * 8][jj * 16 + lmod] = s[jj][r] * invT;
    __syncthreads();

    // ---- online softmax (lanes 0..15 each own one query row)
    if (lane < 16) {
      float tmax = -1e30f;
#pragma unroll
      for (int c = 0; c < 32; ++c) tmax = fmaxf(tmax, sS[lane][c]);
      const float nm = fmaxf(runmax, tmax);
      const float sc = __expf(runmax - nm);
      float ps = 0.0f;
#pragma unroll
      for (int c = 0; c < 32; ++c) {
        const float p = __expf(sS[lane][c] - nm);
        sP[lane][c] = (_Float16)p;
        ps += p;
      }
      runsum = runsum * sc + ps;
      runmax = nm;
      sScale[lane] = sc;
    }
    __syncthreads();

#pragma unroll
    for (int r = 0; r < 8; ++r) {
      const float sc = sScale[r + lhi * 8];
#pragma unroll
      for (int d = 0; d < 8; ++d) o[d][r] *= sc;
    }

    const h16 pf = pack16(*(const h8*)(&sP[lmod][lhi * 16]),
                          *(const h8*)(&sP[lmod][lhi * 16 + 8]));
#pragma unroll
    for (int d = 0; d < 8; ++d)
      o[d] = __builtin_amdgcn_wmma_f32_16x16x32_f16(false, pf, false, vf[d],
                                                    (short)0, o[d], false, false);
    __syncthreads();
  }

  if (lane < 16) sSum[lane] = runsum;
  __syncthreads();
#pragma unroll
  for (int r = 0; r < 8; ++r) {
    const int b = m0 + r + lhi * 8;
    const float inv = 1.0f / sSum[r + lhi * 8];
#pragma unroll
    for (int d = 0; d < 8; ++d)
      att[((size_t)b * TSEQ + t) * CC + hh * DH + d * 16 + lmod] = o[d][r] * inv;
  }
}

// ---------------------------------------------------------------------------
// final: temporal attention pooling + decode. one block per batch element
// ---------------------------------------------------------------------------
__global__ __launch_bounds__(256)
void final_kernel(const float* __restrict__ h, const float* __restrict__ ht,
                  const float* __restrict__ decw, const float* __restrict__ decb,
                  float* __restrict__ out)
{
  __shared__ float red[256];
  __shared__ float sDot[TSEQ];
  __shared__ float sD2[TSEQ];
  const int b = blockIdx.x;
  const int c = threadIdx.x;
  const size_t base = (size_t)b * TSEQ * CC + c;
  const float hl = ht[base + 7 * CC];
  const float dw = decw[c];

  for (int t = 0; t < TSEQ; ++t) {
    red[c] = ht[base + (size_t)t * CC] * hl;
    __syncthreads();
    for (int s = 128; s > 0; s >>= 1) { if (c < s) red[c] += red[c + s]; __syncthreads(); }
    if (c == 0) sDot[t] = red[0];
    __syncthreads();
    red[c] = h[base + (size_t)t * CC] * dw;
    __syncthreads();
    for (int s = 128; s > 0; s >>= 1) { if (c < s) red[c] += red[c + s]; __syncthreads(); }
    if (c == 0) sD2[t] = red[0];
    __syncthreads();
  }
  if (c == 0) {
    float mx = -1e30f;
    for (int t = 0; t < TSEQ; ++t) mx = fmaxf(mx, sDot[t]);
    float s = 0.0f, e[TSEQ];
    for (int t = 0; t < TSEQ; ++t) { e[t] = __expf(sDot[t] - mx); s += e[t]; }
    float acc = 0.0f;
    for (int t = 0; t < TSEQ; ++t) acc += (e[t] / s) * sD2[t];
    out[b] = acc + decb[0];
  }
}

// ---------------------------------------------------------------------------
// launch
// ---------------------------------------------------------------------------
extern "C" void kernel_launch(void* const* d_in, const int* in_sizes, int n_in,
                              void* d_out, int out_size, void* d_ws, size_t ws_size,
                              hipStream_t stream)
{
  (void)in_sizes; (void)n_in; (void)out_size; (void)ws_size;

  const float* X = (const float*)d_in[0];
  // param indices (setup_inputs insertion order)
  const float* FT_W = (const float*)d_in[1];  const float* FT_B = (const float*)d_in[2];
  const float* FC_W = (const float*)d_in[3];  const float* FC_B = (const float*)d_in[4];
  const float* FP_W = (const float*)d_in[5];  const float* FP_B = (const float*)d_in[6];
  const float* IN_W = (const float*)d_in[7];  const float* IN_B = (const float*)d_in[8];
  const float* DLS_W = (const float*)d_in[9]; const float* DLS_B = (const float*)d_in[10];
  const float* DLT_W = (const float*)d_in[11];const float* DLT_B = (const float*)d_in[12];
  const float* LN1G = (const float*)d_in[13]; const float* LN1B = (const float*)d_in[14];
  const float* LN2G = (const float*)d_in[15]; const float* LN2B = (const float*)d_in[16];
  const float* TIME_W = (const float*)d_in[17];
  const float* TIME_A = (const float*)d_in[18];
  const float* TIME_Bp = (const float*)d_in[19];
  const float* TIME_G = (const float*)d_in[20];
  const float* TMK_B = (const float*)d_in[22];
  const float* TMV_B = (const float*)d_in[24];
  const float* TMR_B = (const float*)d_in[26];
  const float* TMO_B = (const float*)d_in[28];
  const float* CMK_B = (const float*)d_in[30];
  const float* CMV_B = (const float*)d_in[32];
  const float* CMR_B = (const float*)d_in[34];
  const float* CMW_B = (const float*)d_in[36];
  const float* SALN1G = (const float*)d_in[37]; const float* SALN1B = (const float*)d_in[38];
  const float* SALN2G = (const float*)d_in[42]; const float* SALN2B = (const float*)d_in[43];
  const float* FF1B = (const float*)d_in[45];
  const float* FF2B = (const float*)d_in[47];
  const float* DECW = (const float*)d_in[49]; const float* DECB = (const float*)d_in[50];

  // workspace layout
  char* ws = (char*)d_ws;
  size_t off = 0;
  auto alloc = [&](size_t bytes) -> char* {
    char* p = ws + off;
    off += (bytes + 255) & ~(size_t)255;
    return p;
  };
  _Float16* W16 = (_Float16*)alloc((size_t)NW * 65536 * sizeof(_Float16));
  float* hbuf = (float*)alloc(MC * sizeof(float));
  float* t0   = (float*)alloc(MC * sizeof(float));
  float* g1   = (float*)alloc(MC * sizeof(float));
  float* g2   = (float*)alloc(MC * sizeof(float));
  float* g3   = (float*)alloc(MC * sizeof(float));
  float* g4   = (float*)alloc(MC * sizeof(float));
  _Float16* x16 = (_Float16*)alloc(MC * sizeof(_Float16));
  _Float16* q16 = (_Float16*)alloc(MC * sizeof(_Float16));
  _Float16* k16 = (_Float16*)alloc(MC * sizeof(_Float16));
  _Float16* v16 = (_Float16*)alloc(MC * sizeof(_Float16));   // holds VT layout

  // convert the 14 256x256 weights to f16
  const int widx[NW] = {21,23,25,27,29,31,33,35,39,40,41,44,46,48};
  WPtrs wp;
  for (int i = 0; i < NW; ++i) wp.p[i] = (const float*)d_in[widx[i]];
  wcvt_kernel<<<NW * 65536 / 256, 256, 0, stream>>>(wp, W16);
  _Float16* Wm[NW];
  for (int i = 0; i < NW; ++i) Wm[i] = W16 + (size_t)i * 65536;
  // 0:tm_k 1:tm_v 2:tm_r 3:tm_o 4:cm_k 5:cm_v 6:cm_r 7:cm_w 8:q 9:k 10:v 11:ffn1 12:ffn2 13:ta

  const int EW = (int)(MC / 256);        // elementwise grid
  const dim3 GG(MM / 16, 4);             // gemm grid (1024 x 4), wave per block
  const dim3 FG(BN / 16, 16);            // flash grid (128 q-tiles x 16 (t,h))

  // ---- prologue -> hbuf
  prologue_kernel<<<BN, 256, 0, stream>>>(X, FT_W, FT_B, FC_W, FC_B, FP_W, FP_B,
                                          IN_W, IN_B, DLS_W, DLS_B, DLT_W, DLT_B, hbuf);

  // ---- timemix
  layernorm_kernel<<<MM, 256, 0, stream>>>(hbuf, nullptr, LN1G, LN1B, t0);
  tshift_cvt_kernel<<<EW, 256, 0, stream>>>(t0, x16);
  gemm256_kernel<1><<<GG, 32, 0, stream>>>(x16, Wm[0], TMK_B, g1, nullptr, 0);  // k=exp(min(.,30))
  gemm256_kernel<0><<<GG, 32, 0, stream>>>(x16, Wm[1], TMV_B, g2, nullptr, 0);  // v
  gemm256_kernel<0><<<GG, 32, 0, stream>>>(x16, Wm[2], TMR_B, g3, nullptr, 0);  // r
  timemix_post_kernel<<<BN, 256, 0, stream>>>(g1, g2, g3, TIME_W, TIME_A, TIME_Bp, q16);
  gemm256_kernel<0><<<GG, 32, 0, stream>>>(q16, Wm[3], TMO_B, g4, nullptr, 0);  // tm_o
  resid_gamma_kernel<<<EW, 256, 0, stream>>>(hbuf, g4, TIME_G);

  // ---- chanmix
  layernorm_kernel<<<MM, 256, 0, stream>>>(hbuf, nullptr, LN2G, LN2B, t0);
  tshift_cvt_kernel<<<EW, 256, 0, stream>>>(t0, x16);
  gemm256_kernel<0><<<GG, 32, 0, stream>>>(x16, Wm[4], CMK_B, g1, nullptr, 0);
  gemm256_kernel<0><<<GG, 32, 0, stream>>>(x16, Wm[5], CMV_B, g2, nullptr, 0);
  gemm256_kernel<0><<<GG, 32, 0, stream>>>(x16, Wm[6], CMR_B, g3, nullptr, 0);
  mishmul_kernel<<<EW, 256, 0, stream>>>(g1, g2, q16);
  gemm256_kernel<0><<<GG, 32, 0, stream>>>(q16, Wm[7], CMW_B, g4, nullptr, 0);
  chanmix_resid_kernel<<<EW, 256, 0, stream>>>(hbuf, g3, g4);

  // ---- batch self-attention
  layernorm_kernel<<<MM, 256, 0, stream>>>(hbuf, nullptr, SALN1G, SALN1B, t0);  // t0 = xs
  cvt_kernel<<<EW, 256, 0, stream>>>(t0, x16);
  gemm256_kernel<0><<<GG, 32, 0, stream>>>(x16, Wm[8],  nullptr, nullptr, q16, 0);
  gemm256_kernel<0><<<GG, 32, 0, stream>>>(x16, Wm[9],  nullptr, nullptr, k16, 0);
  gemm256_kernel<0><<<GG, 32, 0, stream>>>(x16, Wm[10], nullptr, nullptr, v16, 1);  // transposed V
  flash_attn_kernel<<<FG, 32, 0, stream>>>(q16, k16, v16, g1);                      // g1 = att

  // ---- post-attention FFN
  layernorm_kernel<<<MM, 256, 0, stream>>>(g1, t0, SALN2G, SALN2B, g2);  // xt = ln(xs+att)
  cvt_kernel<<<EW, 256, 0, stream>>>(g2, x16);
  gemm256_kernel<2><<<GG, 32, 0, stream>>>(x16, Wm[11], FF1B, nullptr, q16, 0);  // relu
  gemm256_kernel<0><<<GG, 32, 0, stream>>>(q16, Wm[12], FF2B, g3, nullptr, 0);
  add_kernel<<<EW, 256, 0, stream>>>(g2, g3, hbuf);                              // h = xt + ff

  // ---- temporal pooling + decode
  cvt_kernel<<<EW, 256, 0, stream>>>(hbuf, x16);
  gemm256_kernel<0><<<GG, 32, 0, stream>>>(x16, Wm[13], nullptr, g1, nullptr, 0);  // ht
  final_kernel<<<BN, 256, 0, stream>>>(hbuf, g1, DECW, DECB, (float*)d_out);
}
